// NeuralEmbeddingLayer_89807766159445
// MI455X (gfx1250) — matrix operand order repaired and checked by reading, not compiled, not run on hardware
//
#include <hip/hip_runtime.h>

typedef __bf16 bf16;
typedef __attribute__((ext_vector_type(16))) __bf16 v16bf;
typedef __attribute__((ext_vector_type(8)))  __bf16 v8bf;
typedef __attribute__((ext_vector_type(8)))  float   v8f;
typedef __attribute__((ext_vector_type(4)))  float   v4f;

#define B_    32
#define T_    1024
#define C_    512
#define H_    1024
#define SCALE 32.0f   // sqrt(H) = sqrt(1024)

#define BM 128
#define BN 128
#define BK 32
#define NK (C_ / BK)       // 16 K-steps
#define LDK (BK + 8)       // pad to 40 bf16 (80B) per row: 16B-aligned, bank-conflict break

union Frag { v16bf v; v8bf h[2]; };

__global__ __launch_bounds__(256)
void embed_bf16_wmma_kernel(const float* __restrict__ features,      // [B,T,C]
                            const int*   __restrict__ ts,            // [B,T]
                            const int*   __restrict__ date_idx,      // [B]
                            const float* __restrict__ Wtab,          // [64,H,C]
                            const float* __restrict__ bias_tab,      // [64]
                            const float* __restrict__ pos,           // [2048,H]
                            float*       __restrict__ out)           // [B,T,H]
{
    __shared__ bf16 As[2][BM][LDK];
    __shared__ bf16 Bs[2][BN][LDK];

    const int tid = threadIdx.x;
    const int blk = blockIdx.x;
    const int tilesM = T_ / BM;                 // 8
    const int tilesN = H_ / BN;                 // 8
    const int b  = blk / (tilesM * tilesN);
    const int tm = (blk / tilesN) % tilesM;
    const int tn = blk % tilesN;

    const int   date = date_idx[b];             // uniform -> scalar load
    const float bias = bias_tab[date];

    const float* Ag = features + ((size_t)b * T_ + (size_t)tm * BM) * C_;
    const float* Bg = Wtab     + ((size_t)date * H_ + (size_t)tn * BN) * C_;

    const int wave  = tid >> 5;                 // 0..7
    const int lane  = tid & 31;
    const int waveM = wave >> 2;                // 0..1 -> 64 rows each
    const int waveN = wave & 3;                 // 0..3 -> 32 cols each
    const int lmod  = lane & 15;
    const int lhalf = lane >> 4;

    // staging: each thread owns one 16-float row segment of A and of B
    const int rowL = tid >> 1;                  // 0..127
    const int segL = (tid & 1) * 16;            // 0 or 16

    v8f acc[4][2];
#pragma unroll
    for (int i = 0; i < 4; ++i)
#pragma unroll
        for (int j = 0; j < 2; ++j)
            acc[i][j] = (v8f)0.0f;

    float regA[16], regB[16];

    auto gload = [&](int kk) {
        const float* pa = Ag + (size_t)rowL * C_ + kk * BK + segL;
        const float* pb = Bg + (size_t)rowL * C_ + kk * BK + segL;
#pragma unroll
        for (int i = 0; i < 4; ++i) {
            v4f a = *(const v4f*)(pa + i * 4);   // global_load_b128
            v4f w = *(const v4f*)(pb + i * 4);
#pragma unroll
            for (int j = 0; j < 4; ++j) { regA[i*4+j] = a[j]; regB[i*4+j] = w[j]; }
        }
    };

    auto lwrite = [&](int buf) {
        v8bf pa[2], pb[2];
#pragma unroll
        for (int h = 0; h < 2; ++h)
#pragma unroll
            for (int i = 0; i < 8; ++i) {
                pa[h][i] = (bf16)regA[h * 8 + i];
                pb[h][i] = (bf16)regB[h * 8 + i];
            }
        *(v8bf*)&As[buf][rowL][segL]     = pa[0];   // ds_store_b128
        *(v8bf*)&As[buf][rowL][segL + 8] = pa[1];
        *(v8bf*)&Bs[buf][rowL][segL]     = pb[0];
        *(v8bf*)&Bs[buf][rowL][segL + 8] = pb[1];
    };

    auto compute = [&](int buf) {
        Frag af[4], bf[2];
        // A 16x32 bf16 layout (ISA 7.12.2): lane M = lane%16;
        // lanes 0-15: K 0-7 in V0-3, K 16-23 in V4-7; lanes 16-31: K 8-15, K 24-31
#pragma unroll
        for (int mt = 0; mt < 4; ++mt) {
            const bf16* p = &As[buf][waveM * 64 + mt * 16 + lmod][0];
            af[mt].h[0] = *(const v8bf*)(p + lhalf * 8);        // ds_load_b128
            af[mt].h[1] = *(const v8bf*)(p + 16 + lhalf * 8);
        }
        // B 32x16 bf16: lane N = lane%16; lanes 0-15: K 0-7, K 16-23; lanes 16-31: K 8-15, K 24-31
#pragma unroll
        for (int nt = 0; nt < 2; ++nt) {
            const bf16* p = &Bs[buf][waveN * 32 + nt * 16 + lmod][0];
            bf[nt].h[0] = *(const v8bf*)(p + lhalf * 8);
            bf[nt].h[1] = *(const v8bf*)(p + 16 + lhalf * 8);
        }
#pragma unroll
        for (int mt = 0; mt < 4; ++mt)
#pragma unroll
            for (int nt = 0; nt < 2; ++nt)
                acc[mt][nt] = __builtin_amdgcn_wmma_f32_16x16x32_bf16(
                    false, af[mt].v, false, bf[nt].v,
                    (short)0, acc[mt][nt], false, false);
    };

    // double-buffered pipeline: global load (k+1) overlaps WMMA on k
    gload(0);
    lwrite(0);
    __syncthreads();

#pragma unroll 2
    for (int kk = 0; kk < NK; ++kk) {
        if (kk + 1 < NK) gload(kk + 1);
        compute(kk & 1);
        if (kk + 1 < NK) {
            lwrite((kk + 1) & 1);   // other buffer; prior readers passed last barrier
            __syncthreads();
        }
    }

    // fused epilogue: (acc + bias) * sqrt(H) + embed_pos[ts[b,t], h]
    const int* tsb = ts + (size_t)b * T_;
#pragma unroll
    for (int mt = 0; mt < 4; ++mt) {
#pragma unroll
        for (int r = 0; r < 8; ++r) {
            const int t   = tm * BM + waveM * 64 + mt * 16 + lhalf * 8 + r;
            const int tsv = tsb[t];
            const float* posrow = pos + (size_t)tsv * H_;
            float* orow = out + ((size_t)b * T_ + t) * H_;
#pragma unroll
            for (int nt = 0; nt < 2; ++nt) {
                const int h = tn * BN + waveN * 32 + nt * 16 + lmod;
                orow[h] = (acc[mt][nt][r] + bias) * SCALE + posrow[h];
            }
        }
    }
}

extern "C" void kernel_launch(void* const* d_in, const int* in_sizes, int n_in,
                              void* d_out, int out_size, void* d_ws, size_t ws_size,
                              hipStream_t stream) {
    const float* features = (const float*)d_in[0];   // [32,1024,512]
    const int*   ts       = (const int*)  d_in[1];   // [32,1024]
    const int*   date_idx = (const int*)  d_in[2];   // [32]
    const float* spikes   = (const float*)d_in[3];   // [64,1024,512]
    const float* sp_bias  = (const float*)d_in[4];   // [64]
    const float* pos      = (const float*)d_in[5];   // [2048,1024]
    float*       out      = (float*)d_out;           // [32,1024,1024]

    const int grid = B_ * (T_ / BM) * (H_ / BN);     // 32*8*8 = 2048 blocks
    embed_bf16_wmma_kernel<<<grid, 256, 0, stream>>>(
        features, ts, date_idx, spikes, sp_bias, pos, out);
}
